// MVPNet3D_39548058862069
// MI455X (gfx1250) — matrix operand order repaired and checked by reading, not compiled, not run on hardware
//
#include <hip/hip_runtime.h>
#include <hip/hip_bf16.h>
#include <math.h>

#define NPTS   8192
#define BATCH  4
#define NC_TOT (BATCH * NPTS)   // 32768 columns

typedef _Float16 v16h  __attribute__((ext_vector_type(16)));
typedef _Float16 h8    __attribute__((ext_vector_type(8)));
typedef float    f32x8 __attribute__((ext_vector_type(8)));

union H16 { h8 h[2]; v16h v; };

// ---------------------------------------------------------------------------
// Weight pack: f32 [O][Cin] -> f16 [O][ldk] zero-padded; pn2mode reorders the
// deep_in concat (points first in reference) so fused features land at col 0.
// ---------------------------------------------------------------------------
__global__ void pack_weight(const float* __restrict__ src, _Float16* __restrict__ dst,
                            int O, int Cin, int ldk, int pn2mode) {
    int idx = blockIdx.x * blockDim.x + threadIdx.x;
    if (idx >= O * ldk) return;
    int c = idx % ldk, o = idx / ldk;
    int sc;
    if (pn2mode) sc = (c < 64) ? (c + 3) : ((c < 67) ? (c - 64) : -1);
    else         sc = (c < Cin) ? c : -1;
    float v = (sc >= 0) ? src[o * Cin + sc] : 0.0f;
    dst[(size_t)o * ldk + c] = (_Float16)v;
}

// Xt_real[n][224] = concat(points[3], knnpoints.reshape(192)) as f16
__global__ void pack_real(const float* __restrict__ pts, const float* __restrict__ knn,
                          _Float16* __restrict__ dst) {
    int idx = blockIdx.x * blockDim.x + threadIdx.x;
    if (idx >= NC_TOT * 224) return;
    int c = idx % 224, nc = idx / 224;
    int b = nc >> 13, n = nc & 8191;
    float v = 0.0f;
    if (c < 3)        v = pts[(size_t)(b * 3 + c) * NPTS + n];
    else if (c < 195) {
        int cc = c - 3, d = cc >> 6, k = cc & 63;
        v = knn[((size_t)(b * 3 + d) * 64 + k) * NPTS + n];
    }
    dst[(size_t)nc * 224 + c] = (_Float16)v;
}

// Xt_img[n][224] = concat(points[3], imageneighbors permuted .reshape(192))
__global__ void pack_img(const float* __restrict__ pts, const float* __restrict__ img,
                         _Float16* __restrict__ dst) {
    int idx = blockIdx.x * blockDim.x + threadIdx.x;
    if (idx >= NC_TOT * 224) return;
    int c = idx % 224, nc = idx / 224;
    int b = nc >> 13, n = nc & 8191;
    float v = 0.0f;
    if (c < 3)        v = pts[(size_t)(b * 3 + c) * NPTS + n];
    else if (c < 195) {
        int cc = c - 3, d = cc >> 6, k = cc & 63;
        v = img[((size_t)(b * 3 + d) * NPTS + n) * 64 + k];
    }
    dst[(size_t)nc * 224 + c] = (_Float16)v;
}

// Xt_pn2[n][64..95]: points at 64..66 (weight cols reordered to match), pad 0
__global__ void pack_pn2extra(const float* __restrict__ pts, _Float16* __restrict__ dst) {
    int idx = blockIdx.x * blockDim.x + threadIdx.x;
    if (idx >= NC_TOT * 32) return;
    int c = 64 + (idx & 31), nc = idx >> 5;
    int b = nc >> 13, n = nc & 8191;
    float v = (c < 67) ? pts[(size_t)(b * 3 + (c - 64)) * NPTS + n] : 0.0f;
    dst[(size_t)nc * 96 + c] = (_Float16)v;
}

// ---------------------------------------------------------------------------
// min over 64x64 pairwise sq-dists per point -> geo = exp(-min/fw1) + fw2
// one wave32 per point; image neighbors staged in LDS; shfl-xor min reduce
// ---------------------------------------------------------------------------
__global__ void mindist_geo(const float* __restrict__ knn, const float* __restrict__ img,
                            const float* __restrict__ fw1, const float* __restrict__ fw2,
                            float* __restrict__ geo) {
    __shared__ float simg[8][64][3];
    int wid  = threadIdx.x >> 5;
    int lane = threadIdx.x & 31;
    int p = blockIdx.x * 8 + wid;
    int b = p >> 13, n = p & 8191;
    for (int j = lane; j < 192; j += 32) {
        int d = j >> 6, k = j & 63;
        simg[wid][k][d] = img[((size_t)(b * 3 + d) * NPTS + n) * 64 + k];
    }
    __syncthreads();
    float rx0[3], rx1[3];
    for (int d = 0; d < 3; ++d) {
        rx0[d] = knn[((size_t)(b * 3 + d) * 64 + lane) * NPTS + n];
        rx1[d] = knn[((size_t)(b * 3 + d) * 64 + lane + 32) * NPTS + n];
    }
    float m = 3.4e38f;
    #pragma unroll 4
    for (int l = 0; l < 64; ++l) {
        float ix = simg[wid][l][0], iy = simg[wid][l][1], iz = simg[wid][l][2];
        float dx = rx0[0] - ix, dy = rx0[1] - iy, dz = rx0[2] - iz;
        float d0 = dx * dx + dy * dy + dz * dz;
        dx = rx1[0] - ix; dy = rx1[1] - iy; dz = rx1[2] - iz;
        float d1 = dx * dx + dy * dy + dz * dz;
        m = fminf(m, fminf(d0, d1));
    }
    for (int off = 16; off > 0; off >>= 1)
        m = fminf(m, __shfl_xor(m, off, 32));
    if (lane == 0) geo[p] = expf(-(1.0f / fw1[0]) * m) + fw2[0];
}

// ---------------------------------------------------------------------------
// WMMA GEMM: D[o,n] = act( (W·X + bias) * g + bt ).  One wave per 16x16 tile.
//   Xt: f16 [NC_TOT][ldx] (channel-contiguous per point), W: f16 [O][Kp].
// Epilogue can write: f32 [b][O][NPTS], f32 transposed [n][ldoT],
//                     f16 transposed [n][ldo] (feeds the next GEMM).
// ---------------------------------------------------------------------------
__global__ void gemm_wmma(const _Float16* __restrict__ Xt, int ldx, int Kp,
                          const _Float16* __restrict__ W,
                          const float* __restrict__ bias,
                          const float* __restrict__ gamma,
                          const float* __restrict__ beta,
                          int O, int relu,
                          float* __restrict__ outBON,
                          float* __restrict__ outT, int ldoT,
                          _Float16* __restrict__ outXt, int ldo) {
    int lane = threadIdx.x;          // blockDim.x == 32, no divergence -> EXEC all 1s
    int hi = lane >> 4, lo = lane & 15;
    int n0 = blockIdx.x << 4;
    int o0 = blockIdx.y << 4;
    const _Float16* wrow = W  + (size_t)(o0 + lo) * Kp;
    const _Float16* xcol = Xt + (size_t)(n0 + lo) * ldx;
    f32x8 acc = {0.f, 0.f, 0.f, 0.f, 0.f, 0.f, 0.f, 0.f};
    for (int kb = 0; kb < Kp; kb += 32) {
        H16 a, bm;
        a.h[0]  = *(const h8*)(wrow + kb + hi * 8);       // A: K = kb+off .. +7
        a.h[1]  = *(const h8*)(wrow + kb + 16 + hi * 8);  //    K = kb+16+off ..
        bm.h[0] = *(const h8*)(xcol + kb + hi * 8);       // B: K = kb+off .. +7
        bm.h[1] = *(const h8*)(xcol + kb + 16 + hi * 8);  //    K = kb+16+off ..
        acc = __builtin_amdgcn_wmma_f32_16x16x32_f16(
            false, a.v, false, bm.v, (short)0, acc, false, false);
    }
    int colg = n0 + lo;
    int bb = colg >> 13, nn = colg & 8191;
    int rbase = o0 + hi * 8;
    float vals[8];
    #pragma unroll
    for (int r = 0; r < 8; ++r) {
        int row = rbase + r;
        float v = acc[r] + bias[row];
        if (gamma) v = v * gamma[row] + beta[row];
        if (relu)  v = fmaxf(v, 0.0f);
        vals[r] = v;
    }
    if (outBON) {
        #pragma unroll
        for (int r = 0; r < 8; ++r)
            outBON[(size_t)(bb * O + rbase + r) * NPTS + nn] = vals[r];
    }
    if (outT) {
        f32x8 t;
        #pragma unroll
        for (int r = 0; r < 8; ++r) t[r] = vals[r];
        *(f32x8*)(outT + (size_t)colg * ldoT + rbase) = t;
    }
    if (outXt) {
        h8 t;
        #pragma unroll
        for (int r = 0; r < 8; ++r) t[r] = (_Float16)vals[r];
        *(h8*)(outXt + (size_t)colg * ldo + rbase) = t;
    }
}

// cos similarity over 64 channels (transposed f32 layouts), fuse geo, fill deep
__global__ void cos_sim_kernel(const float* __restrict__ sr, const float* __restrict__ si,
                               const float* __restrict__ geo, const float* __restrict__ fw3,
                               float* __restrict__ out_sim, float* __restrict__ out_deep,
                               float* __restrict__ simbuf) {
    int p = blockIdx.x * blockDim.x + threadIdx.x;
    if (p >= NC_TOT) return;
    const float* a = sr + (size_t)p * 64;
    const float* b = si + (size_t)p * 64;
    float num = 0.f, na = 0.f, nb = 0.f;
    #pragma unroll 8
    for (int i = 0; i < 64; ++i) {
        float x = a[i], y = b[i];
        num += x * y; na += x * x; nb += y * y;
    }
    float den = fmaxf(sqrtf(na) * sqrtf(nb), 1e-8f);
    float s = (num / den) * geo[p];
    out_sim[p]  = s;
    out_deep[p] = fw3[0];
    simbuf[p]   = s;
}

// Xt_fused[n][64..127] = f16( sim[n] * feature_2d3d[b,c,n] )
__global__ void fill_f2d3d(const float* __restrict__ simbuf, const float* __restrict__ f2d,
                           _Float16* __restrict__ F) {
    int idx = blockIdx.x * blockDim.x + threadIdx.x;
    if (idx >= NC_TOT * 64) return;
    int c = idx & 63, nc = idx >> 6;
    int b = nc >> 13, n = nc & 8191;
    F[(size_t)nc * 128 + 64 + c] = (_Float16)(simbuf[nc] * f2d[(size_t)(b * 64 + c) * NPTS + n]);
}

// ---------------------------------------------------------------------------
extern "C" void kernel_launch(void* const* d_in, const int* in_sizes, int n_in,
                              void* d_out, int out_size, void* d_ws, size_t ws_size,
                              hipStream_t stream) {
    const float* imgn   = (const float*)d_in[0];
    const float* knn    = (const float*)d_in[1];
    const float* pts    = (const float*)d_in[2];
    const float* f2d    = (const float*)d_in[3];
    const float* sift_W = (const float*)d_in[4];
    const float* sift_b = (const float*)d_in[5];
    const float* sift_g = (const float*)d_in[6];
    const float* sift_bt= (const float*)d_in[7];
    const float* c2_W1  = (const float*)d_in[8];
    const float* c2_b1  = (const float*)d_in[9];
    const float* c2_g1  = (const float*)d_in[10];
    const float* c2_bt1 = (const float*)d_in[11];
    const float* c2_W2  = (const float*)d_in[12];
    const float* c2_b2  = (const float*)d_in[13];
    const float* c2_g2  = (const float*)d_in[14];
    const float* c2_bt2 = (const float*)d_in[15];
    const float* lin1_W = (const float*)d_in[16];
    const float* lin1_b = (const float*)d_in[17];
    const float* c3_W1  = (const float*)d_in[18];
    const float* c3_b1  = (const float*)d_in[19];
    const float* c3_g1  = (const float*)d_in[20];
    const float* c3_bt1 = (const float*)d_in[21];
    const float* c3_W2  = (const float*)d_in[22];
    const float* c3_b2  = (const float*)d_in[23];
    const float* c3_g2  = (const float*)d_in[24];
    const float* c3_bt2 = (const float*)d_in[25];
    const float* pn2_W  = (const float*)d_in[26];
    const float* pn2_b  = (const float*)d_in[27];
    const float* pn2_g  = (const float*)d_in[28];
    const float* pn2_bt = (const float*)d_in[29];
    const float* fw1    = (const float*)d_in[30];
    const float* fw2    = (const float*)d_in[31];
    const float* fw3    = (const float*)d_in[32];

    float* out          = (float*)d_out;
    float* out_sim      = out;
    float* out_deep     = out + NC_TOT;
    float* out_shallow  = out + 2 * NC_TOT;                       // [4][64][8192]
    float* out_deepfeat = out + 2 * NC_TOT + BATCH * 64 * NPTS;   // [4][128][8192]

    char* ws = (char*)d_ws;
    size_t off = 0;
    auto alloc = [&](size_t bytes) -> void* {
        void* p = ws + off;
        off += (bytes + 255) & ~(size_t)255;
        return p;
    };
    _Float16* Wsift = (_Float16*)alloc(128 * 224 * 2);
    _Float16* Wc21  = (_Float16*)alloc(64 * 128 * 2);
    _Float16* Wc22  = (_Float16*)alloc(64 * 64 * 2);
    _Float16* Wlin  = (_Float16*)alloc(64 * 64 * 2);
    _Float16* Wc31  = (_Float16*)alloc(64 * 128 * 2);
    _Float16* Wc32  = (_Float16*)alloc(64 * 64 * 2);
    _Float16* Wpn2  = (_Float16*)alloc(128 * 96 * 2);
    float*    geo   = (float*)alloc((size_t)NC_TOT * 4);
    float*    simb  = (float*)alloc((size_t)NC_TOT * 4);
    _Float16* XtA   = (_Float16*)alloc((size_t)NC_TOT * 224 * 2); // real, then img
    _Float16* T2    = (_Float16*)alloc((size_t)NC_TOT * 128 * 2); // sift out
    _Float16* T3    = (_Float16*)alloc((size_t)NC_TOT * 64 * 2);  // c2_1 / c3_1 out
    _Float16* Ff    = (_Float16*)alloc((size_t)NC_TOT * 128 * 2); // fused (real||f2d3d)
    _Float16* Cimg  = (_Float16*)alloc((size_t)NC_TOT * 64 * 2);  // shallowimagefeat
    float*    R1    = (float*)alloc((size_t)NC_TOT * 64 * 4);     // lin1(real) T
    float*    R2    = (float*)alloc((size_t)NC_TOT * 64 * 4);     // lin1(img) T
    _Float16* XtP   = (_Float16*)alloc((size_t)NC_TOT * 96 * 2);  // pn2 input

    // --- pack weights ---
    pack_weight<<<(128 * 224 + 255) / 256, 256, 0, stream>>>(sift_W, Wsift, 128, 195, 224, 0);
    pack_weight<<<(64 * 128 + 255) / 256, 256, 0, stream>>>(c2_W1, Wc21, 64, 128, 128, 0);
    pack_weight<<<(64 * 64 + 255) / 256, 256, 0, stream>>>(c2_W2, Wc22, 64, 64, 64, 0);
    pack_weight<<<(64 * 64 + 255) / 256, 256, 0, stream>>>(lin1_W, Wlin, 64, 64, 64, 0);
    pack_weight<<<(64 * 128 + 255) / 256, 256, 0, stream>>>(c3_W1, Wc31, 64, 128, 128, 0);
    pack_weight<<<(64 * 64 + 255) / 256, 256, 0, stream>>>(c3_W2, Wc32, 64, 64, 64, 0);
    pack_weight<<<(128 * 96 + 255) / 256, 256, 0, stream>>>(pn2_W, Wpn2, 128, 67, 96, 1);
    pack_pn2extra<<<NC_TOT * 32 / 256, 256, 0, stream>>>(pts, XtP);

    // --- geometric similarity ---
    mindist_geo<<<NC_TOT / 8, 256, 0, stream>>>(knn, imgn, fw1, fw2, geo);

    // --- shallow(real) chain: 195->128->64->64, epilogue stores fused[0..63] ---
    pack_real<<<NC_TOT * 224 / 256, 256, 0, stream>>>(pts, knn, XtA);
    gemm_wmma<<<dim3(NC_TOT / 16, 8), 32, 0, stream>>>(XtA, 224, 224, Wsift,
        sift_b, sift_g, sift_bt, 128, 1, nullptr, nullptr, 0, T2, 128);
    gemm_wmma<<<dim3(NC_TOT / 16, 4), 32, 0, stream>>>(T2, 128, 128, Wc21,
        c2_b1, c2_g1, c2_bt1, 64, 1, nullptr, nullptr, 0, T3, 64);
    gemm_wmma<<<dim3(NC_TOT / 16, 4), 32, 0, stream>>>(T3, 64, 64, Wc22,
        c2_b2, c2_g2, c2_bt2, 64, 1, out_shallow, nullptr, 0, Ff, 128);

    // --- shallow(image) chain (reuses XtA/T2/T3 via stream ordering) ---
    pack_img<<<NC_TOT * 224 / 256, 256, 0, stream>>>(pts, imgn, XtA);
    gemm_wmma<<<dim3(NC_TOT / 16, 8), 32, 0, stream>>>(XtA, 224, 224, Wsift,
        sift_b, sift_g, sift_bt, 128, 1, nullptr, nullptr, 0, T2, 128);
    gemm_wmma<<<dim3(NC_TOT / 16, 4), 32, 0, stream>>>(T2, 128, 128, Wc21,
        c2_b1, c2_g1, c2_bt1, 64, 1, nullptr, nullptr, 0, T3, 64);
    gemm_wmma<<<dim3(NC_TOT / 16, 4), 32, 0, stream>>>(T3, 64, 64, Wc22,
        c2_b2, c2_g2, c2_bt2, 64, 1, nullptr, nullptr, 0, Cimg, 64);

    // --- lin1 (bias only, no BN/ReLU), f32 transposed outputs ---
    gemm_wmma<<<dim3(NC_TOT / 16, 4), 32, 0, stream>>>(Ff, 128, 64, Wlin,
        lin1_b, nullptr, nullptr, 64, 0, nullptr, R1, 64, nullptr, 0);
    gemm_wmma<<<dim3(NC_TOT / 16, 4), 32, 0, stream>>>(Cimg, 64, 64, Wlin,
        lin1_b, nullptr, nullptr, 64, 0, nullptr, R2, 64, nullptr, 0);

    // --- cosine * geo -> shallow_similarity; deep_similarity = fw3 ---
    cos_sim_kernel<<<NC_TOT / 256, 256, 0, stream>>>(R1, R2, geo, fw3,
                                                     out_sim, out_deep, simb);

    // --- fused[64..127] = sim * feature_2d3d ; c3 chain ; pn2 ---
    fill_f2d3d<<<NC_TOT * 64 / 256, 256, 0, stream>>>(simb, f2d, Ff);
    gemm_wmma<<<dim3(NC_TOT / 16, 4), 32, 0, stream>>>(Ff, 128, 128, Wc31,
        c3_b1, c3_g1, c3_bt1, 64, 1, nullptr, nullptr, 0, T3, 64);
    gemm_wmma<<<dim3(NC_TOT / 16, 4), 32, 0, stream>>>(T3, 64, 64, Wc32,
        c3_b2, c3_g2, c3_bt2, 64, 1, nullptr, nullptr, 0, XtP, 96);
    gemm_wmma<<<dim3(NC_TOT / 16, 8), 32, 0, stream>>>(XtP, 96, 96, Wpn2,
        pn2_b, pn2_g, pn2_bt, 128, 1, out_deepfeat, nullptr, 0, nullptr, 0);

    (void)in_sizes; (void)n_in; (void)out_size; (void)ws_size;
}
